// DNN_22222160790042
// MI455X (gfx1250) — matrix operand order repaired and checked by reading, not compiled, not run on hardware
//
#include <hip/hip_runtime.h>

typedef __bf16 bf16;
typedef __attribute__((ext_vector_type(16))) __bf16 v16bf;
typedef __attribute__((ext_vector_type(8)))  float  v8f;

#define NLAYERS 12
// D = [2,10,20,50,100,200,1000,200,100,50,20,10,2]
// Kp = roundup32(Din), Np = roundup32(Dout); Wt stored [Np][Kp] bf16, zero padded.
__device__ __host__ constexpr size_t LWOFF[NLAYERS] =
    {0,1024,2048,4096,12288,40960,270336,499712,528384,536576,538624,539648};
__device__ __host__ constexpr size_t WT_TOTAL = 540672;   // bf16 elems
__device__ __host__ constexpr size_t LBOFF[NLAYERS] =
    {0,32,64,128,256,480,1504,1728,1856,1920,1952,1984};

__device__ inline unsigned short f32_to_bf16_bits(float f) {
  bf16 h = (bf16)f;
  union { bf16 h; unsigned short u; } cv; cv.h = h; return cv.u;
}
__device__ inline float bf16_bits_to_f32(unsigned short u) {
  union { bf16 h; unsigned short u; } cv; cv.u = u; return (float)cv.h;
}

// ---- Prep: W (Din x Dout f32) -> Wt (Np x Kp bf16, transposed, zero pad);
//            bias padded to Np with zeros. -------------------------------
__global__ void prep_weights(const float* __restrict W, const float* __restrict b,
                             unsigned short* __restrict wt, float* __restrict bp,
                             int Din, int Dout, int Kp, int Np) {
  int idx = blockIdx.x * blockDim.x + threadIdx.x;
  int tot = Np * Kp;
  if (idx < tot) {
    int n = idx / Kp;
    int k = idx - n * Kp;
    float v = (n < Dout && k < Din) ? W[(size_t)k * Dout + n] : 0.0f;
    wt[idx] = f32_to_bf16_bits(v);
  }
  if (idx < Np) bp[idx] = (idx < Dout) ? b[idx] : 0.0f;
}

// ---- Fused MLP ----------------------------------------------------------
#define ROWS_PER_BLOCK 128
#define P0 264   // pitch: even activations buffer (max width 256 = L5/L6 chunk)
#define P1 232   // pitch: odd activations buffer (max width 224)

union Frag { uint4 u[2]; v16bf v; };

__device__ inline v8f wmma_bf16(v16bf a, v16bf b, v8f c) {
  return __builtin_amdgcn_wmma_f32_16x16x32_bf16(false, a, false, b,
                                                 (short)0, c, false, false);
}

__device__ inline void store_tile(unsigned short* __restrict aout, int pout,
                                  int rowBase, int mt, int half, int n,
                                  float bv, v8f c) {
  // C layout: VGPR j -> row (mt*16 + j + 8*half), col n
  unsigned short* orow = aout + (size_t)(rowBase + mt * 16 + half * 8) * pout + n;
  #pragma unroll
  for (int j = 0; j < 8; ++j) {
    float v = fmaxf(c[j] + bv, 0.0f);
    orow[(size_t)j * pout] = f32_to_bf16_bits(v);
  }
}

// A-fragment pair (2 M-tiles) loader: row = rowBase + lane%16, K-chunk per layout.
__device__ inline void load_a_pair(const unsigned short* __restrict aRow, int pin,
                                   int ktByteOff, Frag& a0, Frag& a1) {
  const unsigned short* p0 = aRow + ktByteOff;
  a0.u[0] = *(const uint4*)(p0);
  a0.u[1] = *(const uint4*)(p0 + 8);
  const unsigned short* p1 = p0 + (size_t)16 * pin;
  a1.u[0] = *(const uint4*)(p1);
  a1.u[1] = *(const uint4*)(p1 + 8);
}

// Generic layer: nt-outer with all A fragments register-resident (KT <= 7).
template<int KT, int NT, int PIN, int POUT>
__device__ inline void layerA(const unsigned short* __restrict ain,
                              unsigned short* __restrict aout,
                              const unsigned short* __restrict Wt,
                              const float* __restrict bias,
                              int rowBase, int n16, int half) {
  constexpr int Kp = KT * 32;
  Frag a[KT][2];
  const unsigned short* aRow = ain + (size_t)(rowBase + n16) * PIN + half * 16;
  #pragma unroll
  for (int kt = 0; kt < KT; ++kt)
    load_a_pair(aRow, PIN, kt * 32, a[kt][0], a[kt][1]);

  const unsigned short* wBase = Wt + (size_t)n16 * Kp + half * 16;
  for (int nt = 0; nt < NT; ++nt) {
    const float bv = bias[nt * 16 + n16];
    const unsigned short* wRow = wBase + (size_t)nt * 16 * Kp;
    v8f c0 = {}; v8f c1 = {};
    #pragma unroll
    for (int kt = 0; kt < KT; ++kt) {
      Frag b;
      const uint4* wp = (const uint4*)(wRow + kt * 32);
      b.u[0] = wp[0];
      b.u[1] = wp[1];
      c0 = wmma_bf16(a[kt][0].v, b.v, c0);
      c1 = wmma_bf16(a[kt][1].v, b.v, c1);
    }
    store_tile(aout, POUT, rowBase, 0, half, nt * 16 + n16, bv, c0);
    store_tile(aout, POUT, rowBase, 1, half, nt * 16 + n16, bv, c1);
  }
}

// Fused layers 5 (224->1024) and 6 (1024->224): the 1024-wide activation is
// produced and consumed 256 columns at a time through a small chunk buffer.
// L5 input (224 wide) lives entirely in registers (a5[7][2]); L6 accumulators
// (c6[14][2]) stay register-resident across all 4 chunks.
__device__ inline void fused_l5_l6(const unsigned short* __restrict ain,   // buf1: act5
                                   unsigned short* __restrict chunkb,      // buf0: 256-wide chunk
                                   unsigned short* __restrict aout,        // buf1: act7
                                   const unsigned short* __restrict W5,
                                   const float* __restrict b5,
                                   const unsigned short* __restrict W6,
                                   const float* __restrict b6,
                                   int rowBase, int n16, int half) {
  // Cache full L5 input slice for this wave's 32 rows.
  Frag a5[7][2];
  {
    const unsigned short* aRow = ain + (size_t)(rowBase + n16) * P1 + half * 16;
    #pragma unroll
    for (int kt = 0; kt < 7; ++kt)
      load_a_pair(aRow, P1, kt * 32, a5[kt][0], a5[kt][1]);
  }

  v8f c6[14][2];
  #pragma unroll
  for (int nt = 0; nt < 14; ++nt) { c6[nt][0] = (v8f){}; c6[nt][1] = (v8f){}; }

  const unsigned short* w5Base = W5 + (size_t)n16 * 224 + half * 16;
  const unsigned short* w6Base = W6 + (size_t)n16 * 1024 + half * 16;
  const unsigned short* cRow = chunkb + (size_t)(rowBase + n16) * P0 + half * 16;

  for (int ch = 0; ch < 4; ++ch) {
    // ---- L5 partial: produce columns [ch*256, ch*256+256) into chunk buffer.
    for (int t = 0; t < 16; ++t) {
      const int nt = ch * 16 + t;
      const float bv = b5[nt * 16 + n16];
      const unsigned short* wRow = w5Base + (size_t)nt * 16 * 224;
      v8f c0 = {}; v8f c1 = {};
      #pragma unroll
      for (int kt = 0; kt < 7; ++kt) {
        Frag b;
        const uint4* wp = (const uint4*)(wRow + kt * 32);
        b.u[0] = wp[0];
        b.u[1] = wp[1];
        c0 = wmma_bf16(a5[kt][0].v, b.v, c0);
        c1 = wmma_bf16(a5[kt][1].v, b.v, c1);
      }
      store_tile(chunkb, P0, rowBase, 0, half, t * 16 + n16, bv, c0);
      store_tile(chunkb, P0, rowBase, 1, half, t * 16 + n16, bv, c1);
    }
    __syncthreads();
    // ---- L6 partial: consume chunk as 8 K-tiles, accumulate into c6.
    for (int kt6 = 0; kt6 < 8; ++kt6) {
      Frag a0, a1;
      load_a_pair(cRow, P0, kt6 * 32, a0, a1);
      const int ktg = ch * 8 + kt6;        // global K-tile index in W6
      #pragma unroll
      for (int nt = 0; nt < 14; ++nt) {
        Frag b;
        const uint4* wp = (const uint4*)(w6Base + (size_t)nt * 16 * 1024 + ktg * 32);
        b.u[0] = wp[0];
        b.u[1] = wp[1];
        c6[nt][0] = wmma_bf16(a0.v, b.v, c6[nt][0]);
        c6[nt][1] = wmma_bf16(a1.v, b.v, c6[nt][1]);
      }
    }
    __syncthreads();
  }
  // ---- L6 epilogue.
  #pragma unroll
  for (int nt = 0; nt < 14; ++nt) {
    const float bv = b6[nt * 16 + n16];
    store_tile(aout, P1, rowBase, 0, half, nt * 16 + n16, bv, c6[nt][0]);
    store_tile(aout, P1, rowBase, 1, half, nt * 16 + n16, bv, c6[nt][1]);
  }
}

__global__ __launch_bounds__(128)
void mlp_fused(const float* __restrict x,
               const unsigned short* __restrict wsW,
               const float* __restrict wsB,
               float* __restrict out) {
  __shared__ unsigned short buf0[ROWS_PER_BLOCK * P0];  // even acts (max 256)
  __shared__ unsigned short buf1[ROWS_PER_BLOCK * P1];  // odd acts (max 224)

  const int tid  = threadIdx.x;
  const int lane = tid & 31;
  const int wave = tid >> 5;
  const int n16  = lane & 15;
  const int half = lane >> 4;
  const int rowBase = wave * 32;      // wave owns 32 rows (2 M-tiles)
  const size_t blockRow = (size_t)blockIdx.x * ROWS_PER_BLOCK;

  // Input fill: thread t owns row t. cols 0..1 = x (bf16), cols 2..31 = 0.
  {
    float2 v = ((const float2*)x)[blockRow + tid];
    unsigned short* r = &buf0[(size_t)tid * P0];
    r[0] = f32_to_bf16_bits(v.x);
    r[1] = f32_to_bf16_bits(v.y);
    #pragma unroll
    for (int k = 2; k < 32; ++k) r[k] = 0;
  }
  __syncthreads();

  layerA<1, 2, P0, P1>(buf0, buf1, wsW + LWOFF[0],  wsB + LBOFF[0],  rowBase, n16, half);
  __syncthreads();
  layerA<1, 2, P1, P0>(buf1, buf0, wsW + LWOFF[1],  wsB + LBOFF[1],  rowBase, n16, half);
  __syncthreads();
  layerA<1, 4, P0, P1>(buf0, buf1, wsW + LWOFF[2],  wsB + LBOFF[2],  rowBase, n16, half);
  __syncthreads();
  layerA<2, 8, P1, P0>(buf1, buf0, wsW + LWOFF[3],  wsB + LBOFF[3],  rowBase, n16, half);
  __syncthreads();
  layerA<4, 14, P0, P1>(buf0, buf1, wsW + LWOFF[4], wsB + LBOFF[4],  rowBase, n16, half);
  __syncthreads();
  fused_l5_l6(buf1, buf0, buf1,
              wsW + LWOFF[5], wsB + LBOFF[5],
              wsW + LWOFF[6], wsB + LBOFF[6], rowBase, n16, half);
  __syncthreads();
  layerA<7, 8, P1, P0>(buf1, buf0, wsW + LWOFF[7],  wsB + LBOFF[7],  rowBase, n16, half);
  __syncthreads();
  layerA<4, 4, P0, P1>(buf0, buf1, wsW + LWOFF[8],  wsB + LBOFF[8],  rowBase, n16, half);
  __syncthreads();
  layerA<2, 2, P1, P0>(buf1, buf0, wsW + LWOFF[9],  wsB + LBOFF[9],  rowBase, n16, half);
  __syncthreads();
  layerA<1, 2, P0, P1>(buf0, buf1, wsW + LWOFF[10], wsB + LBOFF[10], rowBase, n16, half);
  __syncthreads();
  layerA<1, 2, P1, P0>(buf1, buf0, wsW + LWOFF[11], wsB + LBOFF[11], rowBase, n16, half);
  __syncthreads();

  // Softmax over the 2 live columns; final activations (act12) are in buf0.
  {
    float h0 = bf16_bits_to_f32(buf0[(size_t)tid * P0 + 0]);
    float h1 = bf16_bits_to_f32(buf0[(size_t)tid * P0 + 1]);
    float m  = fmaxf(h0, h1);
    float e0 = __expf(h0 - m), e1 = __expf(h1 - m);
    float s  = e0 + e1;
    float2 r; r.x = e0 / s; r.y = e1 / s;
    ((float2*)out)[blockRow + tid] = r;
  }
}

// ---- Host launch --------------------------------------------------------
extern "C" void kernel_launch(void* const* d_in, const int* in_sizes, int n_in,
                              void* d_out, int out_size, void* d_ws, size_t ws_size,
                              hipStream_t stream) {
  const float* x = (const float*)d_in[0];   // [0]=x, [1..12]=Ws, [13..24]=bs
  unsigned short* wsW = (unsigned short*)d_ws;
  float* wsB = (float*)((char*)d_ws + WT_TOTAL * sizeof(unsigned short));

  static const int hKP[NLAYERS] = {32,32,32,64,128,224,1024,224,128,64,32,32};
  static const int hNP[NLAYERS] = {32,32,64,128,224,1024,224,128,64,32,32,32};
  static const int hDIN[NLAYERS] = {2,10,20,50,100,200,1000,200,100,50,20,10};
  static const int hDOUT[NLAYERS]= {10,20,50,100,200,1000,200,100,50,20,10,2};
  static const size_t hWOFF[NLAYERS] =
      {0,1024,2048,4096,12288,40960,270336,499712,528384,536576,538624,539648};
  static const size_t hBOFF[NLAYERS] =
      {0,32,64,128,256,480,1504,1728,1856,1920,1952,1984};

  for (int L = 0; L < NLAYERS; ++L) {
    int tot = hNP[L] * hKP[L];
    int blocks = (tot + 255) / 256;
    prep_weights<<<blocks, 256, 0, stream>>>(
        (const float*)d_in[1 + L], (const float*)d_in[13 + L],
        wsW + hWOFF[L], wsB + hBOFF[L],
        hDIN[L], hDOUT[L], hKP[L], hNP[L]);
  }

  int nrows = in_sizes[0] / 2;                 // 262144
  int grid  = nrows / ROWS_PER_BLOCK;          // 2048
  mlp_fused<<<grid, ROWS_PER_BLOCK, 0, stream>>>(x, wsW, wsB, (float*)d_out);
}
